// DecoderLayer_18253611008698
// MI455X (gfx1250) — compile-verified, matrix operand order checked by reading, not run on hardware
//
#include <hip/hip_runtime.h>
#include <hip/hip_bf16.h>

typedef __attribute__((ext_vector_type(16))) __bf16 v16bf;
typedef __attribute__((ext_vector_type(8)))  __bf16 v8bf;
typedef __attribute__((ext_vector_type(8)))  float  v8f;

#define B_  4
#define L_  2048
#define E_  512
#define H_  8
#define D_  512
#define F_  2048
#define M_  (B_*L_)   /* 8192 */

// ---------------- WMMA helpers ----------------

__device__ inline v8f wmma_bf16f32(v16bf a, v16bf b, v8f c) {
  // D = A(16x32 bf16) * B(32x16 bf16) + C(16x16 f32)
  return __builtin_amdgcn_wmma_f32_16x16x32_bf16(
      /*neg_a=*/false, a, /*neg_b=*/false, b,
      /*c_mod=*/(short)0, c, /*reuse_a=*/false, /*reuse_b=*/false);
}

__device__ inline v8f zero_v8f() {
  v8f z;
#pragma unroll
  for (int i = 0; i < 8; ++i) z[i] = 0.f;
  return z;
}

// Load a 16x32 bf16 fragment (A layout; B uses the same layout with rows = N)
// from a row-major-over-K tile with element stride `stride`.
// lane l: row = row0 + l%16 ; K = k0 + (l/16)*8 + {0..7} in v[0..7], +16 in v[8..15]
__device__ inline v16bf load_frag_nt(const __bf16* __restrict__ base,
                                     int row0, int k0, int stride) {
  int lane = threadIdx.x & 31;
  const __bf16* p = base + (long)(row0 + (lane & 15)) * stride + k0 + ((lane >> 4) << 3);
  v8bf lo = *reinterpret_cast<const v8bf*>(p);
  v8bf hi = *reinterpret_cast<const v8bf*>(p + 16);
  v16bf f;
#pragma unroll
  for (int i = 0; i < 8; ++i) { f[i] = lo[i]; f[i + 8] = hi[i]; }
  return f;
}

// ---------------- CDNA5 async global->LDS (ASYNCcnt path) ----------------
// Each lane copies 16B straight into LDS, no VGPR staging.
__device__ inline void async_copy_b128(unsigned lds_off, const __bf16* gptr) {
  asm volatile("global_load_async_to_lds_b128 %0, %1, off"
               :: "v"(lds_off), "v"(gptr)
               : "memory");
}
__device__ inline void wait_async0() {
  asm volatile("s_wait_asynccnt 0" ::: "memory");
}

// ---------------- small utility kernels ----------------

__global__ __launch_bounds__(256) void cvt_f32_bf16_kernel(
    const float* __restrict__ x, __bf16* __restrict__ y, int n) {
  int i = blockIdx.x * 256 + threadIdx.x;
  if (i < n) y[i] = (__bf16)x[i];
}

// out[h][c][r] = (bf16) in[h][r][c]   (R rows, C cols per batch z)
__global__ __launch_bounds__(256) void transpose_to_bf16_kernel(
    const float* __restrict__ in, __bf16* __restrict__ out, int R, int C) {
  long hoff = (long)blockIdx.y * R * C;
  int idx = blockIdx.x * 256 + threadIdx.x;
  if (idx < R * C) {
    int r = idx / C, c = idx % C;
    out[hoff + (long)c * R + r] = (__bf16)in[hoff + idx];
  }
}

__global__ __launch_bounds__(256) void zero_f32_kernel(float* __restrict__ p, int n) {
  int i = blockIdx.x * 256 + threadIdx.x;
  if (i < n) p[i] = 0.f;
}

__global__ __launch_bounds__(256) void add_f32_kernel(
    const float* __restrict__ a, const float* __restrict__ b,
    float* __restrict__ o, int n) {
  int i = blockIdx.x * 256 + threadIdx.x;
  if (i < n) o[i] = a[i] + b[i];
}

__global__ __launch_bounds__(256) void add_inplace_f32_kernel(
    float* __restrict__ x, const float* __restrict__ a, int n) {
  int i = blockIdx.x * 256 + threadIdx.x;
  if (i < n) x[i] += a[i];
}

// LayerNorm over E=512; one block (256 thr) per row; bf16 output.
__global__ __launch_bounds__(256) void layernorm_bf16_kernel(
    const float* __restrict__ x, const float* __restrict__ g,
    const float* __restrict__ bta, __bf16* __restrict__ y) {
  int row = blockIdx.x, t = threadIdx.x;
  const float* xr = x + (long)row * E_;
  float v0 = xr[t], v1 = xr[t + 256];
  __shared__ float red[256];
  red[t] = v0 + v1;
  __syncthreads();
#pragma unroll
  for (int s = 128; s > 0; s >>= 1) {
    if (t < s) red[t] += red[t + s];
    __syncthreads();
  }
  float mean = red[0] * (1.0f / E_);
  __syncthreads();
  float d0 = v0 - mean, d1 = v1 - mean;
  red[t] = d0 * d0 + d1 * d1;
  __syncthreads();
#pragma unroll
  for (int s = 128; s > 0; s >>= 1) {
    if (t < s) red[t] += red[t + s];
    __syncthreads();
  }
  float inv = rsqrtf(red[0] * (1.0f / E_) + 1e-5f);
  __bf16* yr = y + (long)row * E_;
  yr[t]       = (__bf16)(d0 * inv * g[t]       + bta[t]);
  yr[t + 256] = (__bf16)(d1 * inv * g[t + 256] + bta[t + 256]);
}

// ---------------- tiled WMMA GEMM:  C[M,N] = A[M,K] * Bt[N,K]^T ----------------
// Double-buffered LDS tiles staged with async-to-LDS; one barrier per K-step.
// mode 0: bf16 store
// mode 1: +bias, relu, bf16 store
// mode 2: +bias +resid, f32 store
// mode 3: bf16 store transposed per (head,batch): C[h][b][n(d)][l] (for V)
__global__ __launch_bounds__(256) void gemm_bf16_nt_kernel(
    const __bf16* __restrict__ A, const __bf16* __restrict__ Bt,
    void* __restrict__ Cout, const float* __restrict__ bias,
    const float* __restrict__ resid,
    int Mdim, int Ndim, int Kdim, int mode,
    long aStride, long bStride, long cStride) {
  __shared__ __attribute__((aligned(16))) __bf16 As[2][128 * 32];
  __shared__ __attribute__((aligned(16))) __bf16 Bs[2][128 * 32];

  int bz = blockIdx.z;
  A  += (long)bz * aStride;
  Bt += (long)bz * bStride;
  int bn0 = blockIdx.x * 128;
  int bm0 = blockIdx.y * 128;
  int t = threadIdx.x, lane = t & 31, w = t >> 5;
  int wm = w & 3;        // 4 M-waves: rows 32*wm
  int wn = w >> 2;       // 2 N-waves: cols 64*wn

  v8f acc[2][4];
#pragma unroll
  for (int i = 0; i < 2; ++i)
#pragma unroll
    for (int j = 0; j < 4; ++j) acc[i][j] = zero_v8f();

  // async stage one 128x32 A-tile + 128x32 B-tile into buffer `buf`
  auto stage = [&](int buf, int k0) {
    const __bf16* Ab = A  + (long)bm0 * Kdim + k0;
    const __bf16* Bb = Bt + (long)bn0 * Kdim + k0;
    unsigned asb = (unsigned)(uintptr_t)(void*)As[buf];
    unsigned bsb = (unsigned)(uintptr_t)(void*)Bs[buf];
#pragma unroll
    for (int p = 0; p < 2; ++p) {
      int c = p * 256 + t;          // 512 chunks of 8 bf16 per tile
      int row = c >> 2, ch = (c & 3) << 3;
      async_copy_b128(asb + (unsigned)(row * 64 + ch * 2),
                      Ab + (long)row * Kdim + ch);
      async_copy_b128(bsb + (unsigned)(row * 64 + ch * 2),
                      Bb + (long)row * Kdim + ch);
      // speculative prefetch of the tile after next (safe: dropped on fault)
      __builtin_prefetch(Ab + (long)row * Kdim + ch + 32);
      __builtin_prefetch(Bb + (long)row * Kdim + ch + 32);
    }
  };

  int nk = Kdim >> 5;
  stage(0, 0);
  for (int kt = 0; kt < nk; ++kt) {
    int cur = kt & 1;
    wait_async0();      // this wave's staging of `cur` complete
    __syncthreads();    // all waves' staging complete
    if (kt + 1 < nk) stage(cur ^ 1, (kt + 1) << 5);  // fill other buffer under compute
    const __bf16* Ac = As[cur];
    const __bf16* Bc = Bs[cur];
    v16bf a0 = load_frag_nt(Ac, 32 * wm, 0, 32);
    v16bf a1 = load_frag_nt(Ac, 32 * wm + 16, 0, 32);
#pragma unroll
    for (int ni = 0; ni < 4; ++ni) {
      v16bf bf = load_frag_nt(Bc, 64 * wn + 16 * ni, 0, 32);
      acc[0][ni] = wmma_bf16f32(a0, bf, acc[0][ni]);
      acc[1][ni] = wmma_bf16f32(a1, bf, acc[1][ni]);
    }
    // no trailing barrier: `cur` is only overwritten after the next iteration's
    // wait+barrier, by which time these ds_loads have been consumed.
  }

#pragma unroll
  for (int mi = 0; mi < 2; ++mi)
#pragma unroll
    for (int ni = 0; ni < 4; ++ni)
#pragma unroll
      for (int r = 0; r < 8; ++r) {
        int m = bm0 + 32 * wm + 16 * mi + r + ((lane >> 4) << 3);
        int n = bn0 + 64 * wn + 16 * ni + (lane & 15);
        float vv = acc[mi][ni][r];
        if (mode == 0) {
          __bf16* Cb = (__bf16*)Cout + (long)bz * cStride;
          Cb[(long)m * Ndim + n] = (__bf16)vv;
        } else if (mode == 1) {
          vv += bias[n];
          vv = vv > 0.f ? vv : 0.f;
          ((__bf16*)Cout)[(long)m * Ndim + n] = (__bf16)vv;
        } else if (mode == 2) {
          vv += bias[n] + resid[(long)m * Ndim + n];
          ((float*)Cout)[(long)m * Ndim + n] = vv;
        } else {
          // V stored transposed: [h][b][n(d)][l]
          __bf16* Cb = (__bf16*)Cout + (long)bz * cStride;
          int bb = m >> 11;            // m / L_
          int ll = m & (L_ - 1);       // m % L_
          Cb[((long)bb * D_ + n) * L_ + ll] = (__bf16)vv;
        }
      }
}

// ---------------- flash attention (64-row blocks, D=512), heads averaged ----------------
// Q/K: bf16 [H][B*L][D].  Vt: bf16 [H][B][D][L] (pre-transposed).
// Oacc: f32 [B*L][E] (pre-zeroed; atomic += headW * O_row/l)
__global__ __launch_bounds__(256) void flash_attn_kernel(
    const __bf16* __restrict__ Q, const __bf16* __restrict__ Kg,
    const __bf16* __restrict__ Vt, float* __restrict__ Oacc,
    int Bn, int Hn, int Ln, float scale, float headW, int causal) {
  __shared__ __attribute__((aligned(16))) __bf16 Ks[64 * 512];   // [key][d]
  __shared__ __attribute__((aligned(16))) __bf16 Vts[512 * 64];  // [d][key]
  __shared__ __attribute__((aligned(16))) float  Ss[64 * 65];    // scores
  __shared__ __attribute__((aligned(16))) __bf16 Ps[64 * 64];    // exp probs
  __shared__ float alpha_s[64];
  __shared__ float rowl[64];

  const int Mtot = Bn * Ln;
  int bm = blockIdx.x, h = blockIdx.y, b = blockIdx.z;
  int t = threadIdx.x, lane = t & 31, w = t >> 5;
  int sm = w & 3, sn = w >> 2;   // S-tile wave grid: rows 16*sm, cols 32*sn
  int m0 = b * Ln + bm * 64;     // global row base into [B*L]
  int q0 = bm * 64;              // sequence-position base

  const unsigned ksb = (unsigned)(uintptr_t)(void*)Ks;   // LDS byte offsets
  const unsigned vsb = (unsigned)(uintptr_t)(void*)Vts;

  // Q fragments resident in registers: rows 16*sm..+15, all K=512
  const __bf16* Qb = Q + ((long)h * Mtot + m0) * D_;
  v16bf qf[16];
#pragma unroll
  for (int kk = 0; kk < 16; ++kk) qf[kk] = load_frag_nt(Qb, 16 * sm, kk * 32, D_);

  // O accumulator: wave w owns cols 64*w, all 64 rows
  v8f of[4][4];
#pragma unroll
  for (int i = 0; i < 4; ++i)
#pragma unroll
    for (int j = 0; j < 4; ++j) of[i][j] = zero_v8f();

  float mrun = -3.0e38f, lrun = 0.f;
  int jmax = causal ? bm : (Ln / 64 - 1);

  for (int j = 0; j <= jmax; ++j) {
    __syncthreads();  // protect Ks/Vts/Ps from previous iteration's readers

    // --- async stage K tile: [64 keys][512 d], contiguous rows ---
    const __bf16* Kb = Kg + ((long)h * Mtot + (long)b * Ln + (long)j * 64) * D_;
#pragma unroll
    for (int p = 0; p < 16; ++p) {      // 4096 x b128
      int c = p * 256 + t;
      int row = c >> 6, col = (c & 63) << 3;
      async_copy_b128(ksb + (unsigned)(row * 1024 + col * 2),
                      Kb + (long)row * D_ + col);
    }
    // --- async stage V^T tile: [512 d][64 keys], 128B contiguous rows ---
    const __bf16* Vb = Vt + ((long)h * Bn + b) * (long)D_ * Ln + (long)j * 64;
#pragma unroll
    for (int p = 0; p < 16; ++p) {      // 4096 x b128
      int c = p * 256 + t;
      int d = c >> 3, ch = (c & 7) << 3;
      async_copy_b128(vsb + (unsigned)(d * 128 + ch * 2),
                      Vb + (long)d * Ln + ch);
    }
    wait_async0();
    __syncthreads();

    // S = Q K^T  (each wave: 16x32 tile)
    v8f s0 = zero_v8f(), s1 = zero_v8f();
#pragma unroll
    for (int kk = 0; kk < 16; ++kk) {
      v16bf b0 = load_frag_nt(Ks, 32 * sn, kk * 32, 512);
      v16bf b1 = load_frag_nt(Ks, 32 * sn + 16, kk * 32, 512);
      s0 = wmma_bf16f32(qf[kk], b0, s0);
      s1 = wmma_bf16f32(qf[kk], b1, s1);
    }
#pragma unroll
    for (int r = 0; r < 8; ++r) {
      int row = 16 * sm + r + ((lane >> 4) << 3);
      Ss[row * 65 + 32 * sn + (lane & 15)]      = s0[r] * scale;
      Ss[row * 65 + 32 * sn + 16 + (lane & 15)] = s1[r] * scale;
    }
    __syncthreads();

    // streaming softmax: thread t < 64 owns row t
    if (t < 64) {
      int qi = q0 + t, kj0 = j * 64;
      float mnew = mrun;
      for (int c = 0; c < 64; ++c) {
        if (causal && (kj0 + c) > qi) continue;
        mnew = fmaxf(mnew, Ss[t * 65 + c]);
      }
      float ladd = 0.f;
      for (int c = 0; c < 64; ++c) {
        float p = (causal && (kj0 + c) > qi) ? 0.f : __expf(Ss[t * 65 + c] - mnew);
        ladd += p;
        Ps[t * 64 + c] = (__bf16)p;
      }
      float al = __expf(mrun - mnew);
      lrun = lrun * al + ladd;
      mrun = mnew;
      alpha_s[t] = al;
    }
    __syncthreads();

    // O = O*alpha + P V  (wave w: cols 64*w)
#pragma unroll
    for (int mi = 0; mi < 4; ++mi)
#pragma unroll
      for (int ni = 0; ni < 4; ++ni)
#pragma unroll
        for (int r = 0; r < 8; ++r) {
          int row = 16 * mi + r + ((lane >> 4) << 3);
          of[mi][ni][r] *= alpha_s[row];
        }
#pragma unroll
    for (int kk = 0; kk < 2; ++kk) {
      v16bf bfr[4];
#pragma unroll
      for (int ni = 0; ni < 4; ++ni)
        bfr[ni] = load_frag_nt(Vts, 64 * w + 16 * ni, 32 * kk, 64);
#pragma unroll
      for (int mi = 0; mi < 4; ++mi) {
        v16bf a = load_frag_nt(Ps, 16 * mi, 32 * kk, 64);
#pragma unroll
        for (int ni = 0; ni < 4; ++ni)
          of[mi][ni] = wmma_bf16f32(a, bfr[ni], of[mi][ni]);
      }
    }
  }

  if (t < 64) rowl[t] = headW / lrun;
  __syncthreads();
#pragma unroll
  for (int mi = 0; mi < 4; ++mi)
#pragma unroll
    for (int ni = 0; ni < 4; ++ni)
#pragma unroll
      for (int r = 0; r < 8; ++r) {
        int row = 16 * mi + r + ((lane >> 4) << 3);
        int gm = m0 + row;
        int gn = 64 * w + 16 * ni + (lane & 15);
        atomicAdd(&Oacc[(long)gm * E_ + gn], of[mi][ni][r] * rowl[row]);
      }
}

// ---------------- orchestration ----------------

extern "C" void kernel_launch(void* const* d_in, const int* in_sizes, int n_in,
                              void* d_out, int out_size, void* d_ws, size_t ws_size,
                              hipStream_t stream) {
  (void)in_sizes; (void)n_in; (void)out_size; (void)ws_size;
  const float* q    = (const float*)d_in[0];
  const float* k    = (const float*)d_in[1];
  const float* v    = (const float*)d_in[2];
  const float* Wq_s = (const float*)d_in[3];
  const float* Wk_s = (const float*)d_in[4];
  const float* Wv_s = (const float*)d_in[5];
  const float* Wq_c = (const float*)d_in[6];
  const float* Wk_c = (const float*)d_in[7];
  const float* Wv_c = (const float*)d_in[8];
  const float* W1   = (const float*)d_in[9];
  const float* b1   = (const float*)d_in[10];
  const float* W2   = (const float*)d_in[11];
  const float* b2   = (const float*)d_in[12];
  const float* g1   = (const float*)d_in[13];
  const float* be1  = (const float*)d_in[14];
  const float* g2   = (const float*)d_in[15];
  const float* be2  = (const float*)d_in[16];
  const float* g3   = (const float*)d_in[17];
  const float* be3  = (const float*)d_in[18];

  char* ws = (char*)d_ws;
  size_t off = 0;
  auto alloc = [&](size_t bytes) -> void* {
    void* p = ws + off;
    off = (off + bytes + 255) & ~(size_t)255;
    return p;
  };

  const size_t wsz = (size_t)H_ * D_ * E_;      // per attn weight tensor (elems)
  __bf16* WqsT = (__bf16*)alloc(wsz * 2);
  __bf16* WksT = (__bf16*)alloc(wsz * 2);
  __bf16* WvsT = (__bf16*)alloc(wsz * 2);
  __bf16* WqcT = (__bf16*)alloc(wsz * 2);
  __bf16* WkcT = (__bf16*)alloc(wsz * 2);
  __bf16* WvcT = (__bf16*)alloc(wsz * 2);
  __bf16* W1T  = (__bf16*)alloc((size_t)F_ * E_ * 2);   // [F][E]
  __bf16* W2T  = (__bf16*)alloc((size_t)E_ * F_ * 2);   // [E][F]
  __bf16* lnb  = (__bf16*)alloc((size_t)M_ * E_ * 2);
  __bf16* knb  = (__bf16*)alloc((size_t)M_ * E_ * 2);
  __bf16* vnb  = (__bf16*)alloc((size_t)M_ * E_ * 2);
  __bf16* Qh   = (__bf16*)alloc((size_t)H_ * M_ * D_ * 2);
  __bf16* Kh   = (__bf16*)alloc((size_t)H_ * M_ * D_ * 2);
  __bf16* Vht  = (__bf16*)alloc((size_t)H_ * M_ * D_ * 2);  // [h][b][d][l]
  float*  acc  = (float*)alloc((size_t)M_ * E_ * 4);
  float*  xbuf = (float*)alloc((size_t)M_ * E_ * 4);
  __bf16* hmid = (__bf16*)alloc((size_t)M_ * F_ * 2);

  dim3 thr(256);
  const int nME = M_ * E_;
  const float scale = 1.0f / sqrtf((float)D_);
  const float headW = 1.0f / (float)H_;

  // weights -> bf16, transposed to [N][K]
  dim3 gWT((E_ * D_ + 255) / 256, H_);
  transpose_to_bf16_kernel<<<gWT, thr, 0, stream>>>(Wq_s, WqsT, E_, D_);
  transpose_to_bf16_kernel<<<gWT, thr, 0, stream>>>(Wk_s, WksT, E_, D_);
  transpose_to_bf16_kernel<<<gWT, thr, 0, stream>>>(Wv_s, WvsT, E_, D_);
  transpose_to_bf16_kernel<<<gWT, thr, 0, stream>>>(Wq_c, WqcT, E_, D_);
  transpose_to_bf16_kernel<<<gWT, thr, 0, stream>>>(Wk_c, WkcT, E_, D_);
  transpose_to_bf16_kernel<<<gWT, thr, 0, stream>>>(Wv_c, WvcT, E_, D_);
  transpose_to_bf16_kernel<<<dim3((E_ * F_ + 255) / 256, 1), thr, 0, stream>>>(W1, W1T, E_, F_);
  transpose_to_bf16_kernel<<<dim3((F_ * E_ + 255) / 256, 1), thr, 0, stream>>>(W2, W2T, F_, E_);
  cvt_f32_bf16_kernel<<<(nME + 255) / 256, thr, 0, stream>>>(k, knb, nME);
  cvt_f32_bf16_kernel<<<(nME + 255) / 256, thr, 0, stream>>>(v, vnb, nME);

  dim3 gProj(D_ / 128, M_ / 128, H_);
  dim3 gAttn(L_ / 64, H_, B_);

  // ---- stage 1: causal self-attention, pre-norm ----
  layernorm_bf16_kernel<<<M_, thr, 0, stream>>>(q, g1, be1, lnb);
  gemm_bf16_nt_kernel<<<gProj, thr, 0, stream>>>(lnb, WqsT, Qh, nullptr, nullptr,
      M_, D_, E_, 0, 0L, (long)D_ * E_, (long)M_ * D_);
  gemm_bf16_nt_kernel<<<gProj, thr, 0, stream>>>(lnb, WksT, Kh, nullptr, nullptr,
      M_, D_, E_, 0, 0L, (long)D_ * E_, (long)M_ * D_);
  gemm_bf16_nt_kernel<<<gProj, thr, 0, stream>>>(lnb, WvsT, Vht, nullptr, nullptr,
      M_, D_, E_, 3, 0L, (long)D_ * E_, (long)M_ * D_);
  zero_f32_kernel<<<(nME + 255) / 256, thr, 0, stream>>>(acc, nME);
  flash_attn_kernel<<<gAttn, thr, 0, stream>>>(Qh, Kh, Vht, acc, B_, H_, L_, scale, headW, 1);
  add_f32_kernel<<<(nME + 255) / 256, thr, 0, stream>>>(q, acc, xbuf, nME);

  // ---- stage 2: cross-attention (queries normed; k,v raw) ----
  layernorm_bf16_kernel<<<M_, thr, 0, stream>>>(xbuf, g2, be2, lnb);
  gemm_bf16_nt_kernel<<<gProj, thr, 0, stream>>>(lnb, WqcT, Qh, nullptr, nullptr,
      M_, D_, E_, 0, 0L, (long)D_ * E_, (long)M_ * D_);
  gemm_bf16_nt_kernel<<<gProj, thr, 0, stream>>>(knb, WkcT, Kh, nullptr, nullptr,
      M_, D_, E_, 0, 0L, (long)D_ * E_, (long)M_ * D_);
  gemm_bf16_nt_kernel<<<gProj, thr, 0, stream>>>(vnb, WvcT, Vht, nullptr, nullptr,
      M_, D_, E_, 3, 0L, (long)D_ * E_, (long)M_ * D_);
  zero_f32_kernel<<<(nME + 255) / 256, thr, 0, stream>>>(acc, nME);
  flash_attn_kernel<<<gAttn, thr, 0, stream>>>(Qh, Kh, Vht, acc, B_, H_, L_, scale, headW, 0);
  add_inplace_f32_kernel<<<(nME + 255) / 256, thr, 0, stream>>>(xbuf, acc, nME);

  // ---- FFN ----
  layernorm_bf16_kernel<<<M_, thr, 0, stream>>>(xbuf, g3, be3, lnb);
  gemm_bf16_nt_kernel<<<dim3(F_ / 128, M_ / 128, 1), thr, 0, stream>>>(
      lnb, W1T, hmid, b1, nullptr, M_, F_, E_, 1, 0L, 0L, 0L);
  gemm_bf16_nt_kernel<<<dim3(E_ / 128, M_ / 128, 1), thr, 0, stream>>>(
      hmid, W2T, d_out, b2, xbuf, M_, E_, F_, 2, 0L, 0L, 0L);
}
